// GPT_15195594293595
// MI455X (gfx1250) — matrix operand order repaired and checked by reading, not compiled, not run on hardware
//
#include <hip/hip_runtime.h>
#include <math.h>

// ---------------------------------------------------------------------------
// GPT-2 small forward for gfx1250 (MI455X). bf16 WMMA, f32 accumulate.
// Activations bf16 (residual stream fp32), async LDS DMA staging, double-
// buffered tiles. B=4 T=1024 D=768 H=12 HD=64 L=12 FF=3072 V=50257
// ---------------------------------------------------------------------------

typedef __attribute__((ext_vector_type(16))) __bf16 bf16x16;
typedef __attribute__((ext_vector_type(8)))  __bf16 bf16x8;
typedef __attribute__((ext_vector_type(8)))  float  f32x8;

constexpr int Bc = 4, Tc = 1024, Dc = 768, Hc = 12, HDc = 64, Lc = 12;
constexpr int Vc = 50257, FFc = 3072;
constexpr int BTc = Bc * Tc;

static __device__ __forceinline__ f32x8 zero8() {
  f32x8 z;
#pragma unroll
  for (int i = 0; i < 8; ++i) z[i] = 0.f;
  return z;
}

static __device__ __forceinline__ f32x8 wmma_bf16(bf16x16 a, bf16x16 b, f32x8 c) {
  return __builtin_amdgcn_wmma_f32_16x16x32_bf16(false, a, false, b, (short)0, c,
                                                 false, false);
}

// A-operand fragment (16x32): lane l15 = row, K = hlf*16 + [0..15] contiguous
// -> two b128 loads
static __device__ __forceinline__ bf16x16 frag_a(const __bf16* p) {
  union { bf16x16 v; bf16x8 h[2]; } u;
  u.h[0] = *reinterpret_cast<const bf16x8*>(p);
  u.h[1] = *reinterpret_cast<const bf16x8*>(p + 8);
  return u.v;
}
// B-operand fragment (32x16): lane l15 = col, K = hlf*16 + [0..15] contiguous
static __device__ __forceinline__ bf16x16 frag_b(const __bf16* p) {
  union { bf16x16 v; bf16x8 h[2]; } u;
  u.h[0] = *reinterpret_cast<const bf16x8*>(p);
  u.h[1] = *reinterpret_cast<const bf16x8*>(p + 8);
  return u.v;
}

// LDS byte offset of a __shared__ object (low 32 bits of the flat address)
static __device__ __forceinline__ unsigned lds_off(const void* p) {
  return (unsigned)(unsigned long long)p;
}
// Async byte copy global -> LDS (16B per lane), tracked by ASYNCcnt.
static __device__ __forceinline__ void async_copy_b128(unsigned lds_byte,
                                                       const void* gsrc) {
  asm volatile("global_load_async_to_lds_b128 %0, %1, off" ::"v"(lds_byte),
               "v"(gsrc)
               : "memory");
}
static __device__ __forceinline__ void wait_async0() {
  asm volatile("s_wait_asynccnt 0x0" ::: "memory");
}

// ---------------------------------------------------------------------------
// Embedding: x[bt, d] = tok_emb[idx[bt], d] + pos_emb[t, d]   (fp32 residual)
// ---------------------------------------------------------------------------
__global__ void gpt_embed_kernel(const int* __restrict__ idx,
                                 const float* __restrict__ tok,
                                 const float* __restrict__ pos,
                                 float* __restrict__ x) {
  const int i = blockIdx.x * 256 + threadIdx.x;
  if (i >= BTc * Dc) return;
  const int d = i % Dc, bt = i / Dc, t = bt % Tc;
  x[i] = tok[(size_t)idx[bt] * Dc + d] + pos[(size_t)t * Dc + d];
}

// ---------------------------------------------------------------------------
// LayerNorm over D=768 (fp32 in, bf16 out), one block per row
// ---------------------------------------------------------------------------
__global__ __launch_bounds__(256) void gpt_ln_kernel(const float* __restrict__ in,
                                                     const float* __restrict__ g,
                                                     const float* __restrict__ bta,
                                                     __bf16* __restrict__ out) {
  __shared__ float red[256];
  const int row = blockIdx.x, tid = threadIdx.x;
  const float* xr = in + (size_t)row * Dc;
  __bf16* yr = out + (size_t)row * Dc;
  float s = 0.f;
#pragma unroll
  for (int it = 0; it < 3; ++it) s += xr[tid + it * 256];
  red[tid] = s;
  __syncthreads();
  for (int k = 128; k > 0; k >>= 1) {
    if (tid < k) red[tid] += red[tid + k];
    __syncthreads();
  }
  const float mean = red[0] * (1.f / Dc);
  __syncthreads();
  float vv = 0.f;
#pragma unroll
  for (int it = 0; it < 3; ++it) {
    const float d = xr[tid + it * 256] - mean;
    vv += d * d;
  }
  red[tid] = vv;
  __syncthreads();
  for (int k = 128; k > 0; k >>= 1) {
    if (tid < k) red[tid] += red[tid + k];
    __syncthreads();
  }
  const float inv = rsqrtf(red[0] * (1.f / Dc) + 1e-5f);
#pragma unroll
  for (int it = 0; it < 3; ++it) {
    const int j = tid + it * 256;
    yr[j] = (__bf16)((xr[j] - mean) * inv * g[j] + bta[j]);
  }
}

// ---------------------------------------------------------------------------
// GEMM: C[M,N] = epilogue(A[M,K](bf16) @ W[K,N](fp32) + bias [+res])
// Block tile 128x128x32, 256 threads = 8 waves (2x4), wave tile 64x32.
// Double-buffered LDS; A tile staged via async LDS DMA (pure byte copy);
// W tile register-staged fp32 -> cvt bf16 -> LDS transposed [n][k].
// kind: 0 plain, 1 +residual(fp32), 2 exact GELU.  obf: 1 -> bf16 output.
// ---------------------------------------------------------------------------
#define BM 128
#define BN 128
#define BKk 32
#define LDAS 40  // BK+8 pad (bf16 units); 80B row stride keeps 16B alignment
#define LDBS 40

__global__ __launch_bounds__(256) void gpt_gemm_kernel(
    const __bf16* __restrict__ A, int lda, const float* __restrict__ W, int ldb,
    const float* __restrict__ bias, const float* __restrict__ res,
    void* __restrict__ Cout, int ldc, int N, int K, int kind, int obf) {
  __shared__ __bf16 Alds[2][BM * LDAS];
  __shared__ __bf16 Blds[2][BN * LDBS];
  const int tid = threadIdx.x;
  const int lane = tid & 31, wid = tid >> 5;
  const int l15 = lane & 15, hlf = lane >> 4;
  const int wm = wid & 1, wn = wid >> 1;
  const int m0 = blockIdx.y * BM;
  const int n0 = blockIdx.x * BN;
  const bool vecW = (n0 + BN <= N) && ((ldb & 3) == 0);

  f32x8 acc[4][2];
#pragma unroll
  for (int i = 0; i < 4; ++i)
#pragma unroll
    for (int j = 0; j < 2; ++j) acc[i][j] = zero8();

  alignas(16) float wreg[16];

  auto stageA = [&](int buf, int k0) {  // 128x32 bf16 = 8KB, 2x16B per thread
    const unsigned base = lds_off(&Alds[buf][0]);
#pragma unroll
    for (int it = 0; it < 2; ++it) {
      const int i = tid + it * 256;  // 512 segments of 8 bf16
      const int m = i >> 2, seg = i & 3;
      async_copy_b128(base + (unsigned)((m * LDAS + seg * 8) * 2),
                      A + (size_t)(m0 + m) * lda + k0 + seg * 8);
    }
  };
  auto loadW = [&](int k0) {  // 32x128 fp32 into 16 regs/thread
    if (vecW) {
#pragma unroll
      for (int it = 0; it < 4; ++it) {
        const int i = tid + it * 256;
        const int k = i >> 5, n4 = (i & 31) * 4;
        *reinterpret_cast<float4*>(&wreg[it * 4]) =
            *reinterpret_cast<const float4*>(W + (size_t)(k0 + k) * ldb + n0 + n4);
      }
    } else {
#pragma unroll
      for (int it = 0; it < 4; ++it) {
        const int i = tid + it * 256;
        const int k = i >> 5, n4 = (i & 31) * 4;
        const float* src = W + (size_t)(k0 + k) * ldb + n0 + n4;
#pragma unroll
        for (int c = 0; c < 4; ++c)
          wreg[it * 4 + c] = (n0 + n4 + c < N) ? src[c] : 0.f;
      }
    }
  };
  auto storeW = [&](int buf) {  // cvt + transposed scatter [n][k]
#pragma unroll
    for (int it = 0; it < 4; ++it) {
      const int i = tid + it * 256;
      const int k = i >> 5, n4 = (i & 31) * 4;
#pragma unroll
      for (int c = 0; c < 4; ++c)
        Blds[buf][(n4 + c) * LDBS + k] = (__bf16)wreg[it * 4 + c];
    }
  };

  // prologue: stage tile 0
  stageA(0, 0);
  loadW(0);
  storeW(0);
  wait_async0();
  __syncthreads();

  const int nk = K / BKk;
  for (int kt = 0; kt < nk; ++kt) {
    const int cur = kt & 1;
    const bool hasNext = (kt + 1 < nk);
    if (hasNext) {            // fire next tile's DMA + global loads now,
      stageA(cur ^ 1, (kt + 1) * BKk);  // overlapped with this tile's WMMAs
      loadW((kt + 1) * BKk);
    }

    bf16x16 af[4], bfr[2];
#pragma unroll
    for (int im = 0; im < 4; ++im)
      af[im] = frag_a(&Alds[cur][(wm * 64 + im * 16 + l15) * LDAS + hlf * 16]);
#pragma unroll
    for (int in = 0; in < 2; ++in)
      bfr[in] = frag_b(&Blds[cur][(wn * 32 + in * 16 + l15) * LDBS + hlf * 16]);
#pragma unroll
    for (int im = 0; im < 4; ++im)
#pragma unroll
      for (int in = 0; in < 2; ++in)
        acc[im][in] = wmma_bf16(af[im], bfr[in], acc[im][in]);

    if (hasNext) storeW(cur ^ 1);
    wait_async0();
    __syncthreads();
  }

  // Epilogue. C layout: lane l15 = col, VGPR r holds row r + 8*hlf.
  float* Cf = (float*)Cout;
  __bf16* Cb = (__bf16*)Cout;
#pragma unroll
  for (int im = 0; im < 4; ++im)
#pragma unroll
    for (int in = 0; in < 2; ++in) {
      const int nn = n0 + wn * 32 + in * 16 + l15;
      if (nn >= N) continue;
      const float bi = bias ? bias[nn] : 0.f;
#pragma unroll
      for (int r = 0; r < 8; ++r) {
        const int mm = m0 + wm * 64 + im * 16 + r + 8 * hlf;
        float v = acc[im][in][r] + bi;
        if (kind == 2) v = 0.5f * v * (1.f + erff(v * 0.70710678118654752f));
        if (kind == 1) v += res[(size_t)mm * ldc + nn];
        if (obf) Cb[(size_t)mm * ldc + nn] = (__bf16)v;
        else     Cf[(size_t)mm * ldc + nn] = v;
      }
    }
}

// ---------------------------------------------------------------------------
// Flash attention (causal), bf16 q/k/v/o. Grid (T/64, B*H), 128 thr = 4 waves.
// Q fragments loaded straight from global (already in WMMA A layout).
// K tiles: async LDS DMA; V tiles: transposed LDS scatter. Double-buffered.
// ---------------------------------------------------------------------------
__global__ __launch_bounds__(128) void gpt_attn_kernel(
    const __bf16* __restrict__ q, const __bf16* __restrict__ k,
    const __bf16* __restrict__ v, __bf16* __restrict__ o) {
  __shared__ __bf16 Klds[2][32 * 72];  // [s][hd], 144B row stride
  __shared__ __bf16 Vt[2][64 * 40];    // [hd][s]
  __shared__ __bf16 Pl[4 * 16 * 40];   // per-wave P re-layout buffer
  const int tid = threadIdx.x;
  const int lane = tid & 31, wid = tid >> 5;
  const int l15 = lane & 15, hlf = lane >> 4;
  const int qblk = blockIdx.x;
  const int b = blockIdx.y / Hc, h = blockIdx.y % Hc;
  const int qbase = qblk * 64 + wid * 16;
  const size_t rowbase = (size_t)b * Tc;
  const float scale = 0.125f;  // 1/sqrt(HD)

  // Q fragments: 16 rows x 64 hd -> two K=32 chunks, direct global b128 loads
  const __bf16* qg = q + (rowbase + qbase + l15) * Dc + h * HDc;
  bf16x16 qf[2];
#pragma unroll
  for (int kc = 0; kc < 2; ++kc) qf[kc] = frag_a(qg + kc * 32 + hlf * 16);

  f32x8 oacc[4];
#pragma unroll
  for (int nt = 0; nt < 4; ++nt) oacc[nt] = zero8();
  float mrow[8], lrow[8];
#pragma unroll
  for (int r = 0; r < 8; ++r) { mrow[r] = -INFINITY; lrow[r] = 0.f; }

  auto stageK = [&](int buf, int sb) {  // 32x64 bf16, async byte copy
    const unsigned base = lds_off(&Klds[buf][0]);
#pragma unroll
    for (int it = 0; it < 2; ++it) {
      const int i = tid + it * 128;  // 256 segments of 8 bf16
      const int s = i >> 3, seg = i & 7;
      async_copy_b128(base + (unsigned)((s * 72 + seg * 8) * 2),
                      k + (rowbase + sb + s) * Dc + h * HDc + seg * 8);
    }
  };
  auto stageV = [&](int buf, int sb) {  // transposed scatter [hd][s]
#pragma unroll
    for (int it = 0; it < 2; ++it) {
      const int i = tid + it * 128;
      const int s = i >> 3, seg = i & 7;
      const bf16x8 d = *reinterpret_cast<const bf16x8*>(
          v + (rowbase + sb + s) * Dc + h * HDc + seg * 8);
#pragma unroll
      for (int j = 0; j < 8; ++j) Vt[buf][(seg * 8 + j) * 40 + s] = d[j];
    }
  };

  __bf16* Pw = &Pl[wid * 16 * 40];
  stageK(0, 0);
  stageV(0, 0);
  wait_async0();
  __syncthreads();

  const int niter = (qblk * 64 + 64) / 32;  // uniform across the 4 waves
  for (int itn = 0; itn < niter; ++itn) {
    const int sb = itn * 32;
    const int cur = itn & 1;
    if (itn + 1 < niter) {  // prefetch next K/V tile into other buffer
      stageK(cur ^ 1, sb + 32);
      stageV(cur ^ 1, sb + 32);
    }

    // S = Q @ K^T for two 16-col s-subtiles
    f32x8 sacc[2];
#pragma unroll
    for (int st = 0; st < 2; ++st) {
      sacc[st] = zero8();
#pragma unroll
      for (int kc = 0; kc < 2; ++kc)
        sacc[st] = wmma_bf16(
            qf[kc],
            frag_b(&Klds[cur][(st * 16 + l15) * 72 + kc * 32 + hlf * 16]),
            sacc[st]);
    }
    // scale + causal mask + online softmax (row = r + 8*hlf, col = l15)
#pragma unroll
    for (int r = 0; r < 8; ++r) {
      const int qrow = qbase + r + 8 * hlf;
#pragma unroll
      for (int st = 0; st < 2; ++st) {
        const int col = sb + st * 16 + l15;
        sacc[st][r] = (col > qrow) ? -INFINITY : sacc[st][r] * scale;
      }
      float mx = fmaxf(sacc[0][r], sacc[1][r]);
      mx = fmaxf(mx, __shfl_xor(mx, 1, 16));
      mx = fmaxf(mx, __shfl_xor(mx, 2, 16));
      mx = fmaxf(mx, __shfl_xor(mx, 4, 16));
      mx = fmaxf(mx, __shfl_xor(mx, 8, 16));
      const float mn = fmaxf(mrow[r], mx);
      const float f = expf(mrow[r] - mn);
      mrow[r] = mn;
      const float p0 = expf(sacc[0][r] - mn);
      const float p1 = expf(sacc[1][r] - mn);
      float rs = p0 + p1;
      rs += __shfl_xor(rs, 1, 16);
      rs += __shfl_xor(rs, 2, 16);
      rs += __shfl_xor(rs, 4, 16);
      rs += __shfl_xor(rs, 8, 16);
      lrow[r] = lrow[r] * f + rs;
#pragma unroll
      for (int nt = 0; nt < 4; ++nt) oacc[nt][r] *= f;
      Pw[(r + 8 * hlf) * 40 + l15] = (__bf16)p0;
      Pw[(r + 8 * hlf) * 40 + 16 + l15] = (__bf16)p1;
    }
    __syncthreads();  // P visible for cross-lane A-layout reads

    // O += P(16x32) @ V(32x64)
    const bf16x16 pf = frag_a(&Pw[l15 * 40 + hlf * 16]);
#pragma unroll
    for (int nt = 0; nt < 4; ++nt)
      oacc[nt] = wmma_bf16(
          pf, frag_b(&Vt[cur][(nt * 16 + l15) * 40 + hlf * 16]), oacc[nt]);

    wait_async0();
    __syncthreads();  // next tile's buffers ready; P/V reads complete
  }

  // normalize and store bf16
#pragma unroll
  for (int r = 0; r < 8; ++r) {
    const float inv = 1.f / lrow[r];
    const size_t row = rowbase + qbase + r + 8 * hlf;
#pragma unroll
    for (int nt = 0; nt < 4; ++nt)
      o[row * Dc + h * HDc + nt * 16 + l15] = (__bf16)(oacc[nt][r] * inv);
  }
}

// ---------------------------------------------------------------------------
// Loss: -mean(log_softmax(logits)[target]); logits untouched.
// ---------------------------------------------------------------------------
__global__ void gpt_zero_kernel(float* p) {
  if (threadIdx.x == 0) p[0] = 0.f;
}

__global__ __launch_bounds__(256) void gpt_loss_kernel(
    const float* __restrict__ logits, const int* __restrict__ tgt,
    float* __restrict__ loss) {
  __shared__ float red[256];
  const int row = blockIdx.x, tid = threadIdx.x;
  const float* lr = logits + (size_t)row * Vc;
  float mx = -INFINITY;
  for (int j = tid; j < Vc; j += 256) mx = fmaxf(mx, lr[j]);
  red[tid] = mx;
  __syncthreads();
  for (int k = 128; k > 0; k >>= 1) {
    if (tid < k) red[tid] = fmaxf(red[tid], red[tid + k]);
    __syncthreads();
  }
  mx = red[0];
  __syncthreads();
  float s = 0.f;
  for (int j = tid; j < Vc; j += 256) s += expf(lr[j] - mx);
  red[tid] = s;
  __syncthreads();
  for (int k = 128; k > 0; k >>= 1) {
    if (tid < k) red[tid] += red[tid + k];
    __syncthreads();
  }
  if (tid == 0) {
    const float lp = lr[tgt[row]] - mx - logf(red[0]);
    atomicAdd(loss, -lp * (1.f / BTc));
  }
}

// ---------------------------------------------------------------------------
// Host orchestration. Workspace: x fp32 (12.6MB) + bf16 h/q/k/v/attn (6.3MB
// each) + bf16 mid (25MB)  => ~69 MB.
// ---------------------------------------------------------------------------
extern "C" void kernel_launch(void* const* d_in, const int* in_sizes, int n_in,
                              void* d_out, int out_size, void* d_ws,
                              size_t ws_size, hipStream_t stream) {
  (void)in_sizes; (void)n_in; (void)out_size; (void)ws_size;
  const int* idx = (const int*)d_in[0];
  const int* tgt = (const int*)d_in[1];
  const float* tok = (const float*)d_in[2];
  const float* pos = (const float*)d_in[3];
  const float* Wq = (const float*)d_in[4];
  const float* Wk = (const float*)d_in[5];
  const float* Wv = (const float*)d_in[6];
  const float* Wo = (const float*)d_in[7];
  const float* bo = (const float*)d_in[8];
  const float* ln1g = (const float*)d_in[9];
  const float* ln1b = (const float*)d_in[10];
  const float* ln2g = (const float*)d_in[11];
  const float* ln2b = (const float*)d_in[12];
  const float* W1 = (const float*)d_in[13];
  const float* b1 = (const float*)d_in[14];
  const float* W2 = (const float*)d_in[15];
  const float* b2 = (const float*)d_in[16];
  const float* lnfg = (const float*)d_in[17];
  const float* lnfb = (const float*)d_in[18];
  const float* Wh = (const float*)d_in[19];
  const float* bh = (const float*)d_in[20];

  float* x = (float*)d_ws;                         // [BT, D] fp32 residual
  __bf16* hb = (__bf16*)(x + (size_t)BTc * Dc);    // LN output
  __bf16* qb = hb + (size_t)BTc * Dc;
  __bf16* kb = qb + (size_t)BTc * Dc;
  __bf16* vb = kb + (size_t)BTc * Dc;
  __bf16* ab = vb + (size_t)BTc * Dc;              // attention output
  __bf16* mid = ab + (size_t)BTc * Dc;             // [BT, FF] GELU output

  float* logits = (float*)d_out;
  float* loss = logits + (size_t)BTc * Vc;

  auto gemm = [&](const __bf16* A, const float* W, const float* bias,
                  const float* res, void* C, int N, int K, int kind, int obf) {
    dim3 g((N + BN - 1) / BN, BTc / BM);
    gpt_gemm_kernel<<<g, 256, 0, stream>>>(A, K, W, N, bias, res, C, N, N, K,
                                           kind, obf);
  };

  gpt_embed_kernel<<<(BTc * Dc + 255) / 256, 256, 0, stream>>>(idx, tok, pos, x);

  for (int l = 0; l < Lc; ++l) {
    gpt_ln_kernel<<<BTc, 256, 0, stream>>>(x, ln1g + l * Dc, ln1b + l * Dc, hb);
    gemm(hb, Wq + (size_t)l * Dc * Dc, nullptr, nullptr, qb, Dc, Dc, 0, 1);
    gemm(hb, Wk + (size_t)l * Dc * Dc, nullptr, nullptr, kb, Dc, Dc, 0, 1);
    gemm(hb, Wv + (size_t)l * Dc * Dc, nullptr, nullptr, vb, Dc, Dc, 0, 1);
    gpt_attn_kernel<<<dim3(Tc / 64, Bc * Hc), 128, 0, stream>>>(qb, kb, vb, ab);
    gemm(ab, Wo + (size_t)l * Dc * Dc, bo + l * Dc, x, x, Dc, Dc, 1, 0);
    gpt_ln_kernel<<<BTc, 256, 0, stream>>>(x, ln2g + l * Dc, ln2b + l * Dc, hb);
    gemm(hb, W1 + (size_t)l * Dc * FFc, b1 + l * FFc, nullptr, mid, FFc, Dc, 2, 1);
    gemm(mid, W2 + (size_t)l * FFc * Dc, b2 + l * Dc, x, x, Dc, FFc, 1, 0);
  }

  gpt_ln_kernel<<<BTc, 256, 0, stream>>>(x, lnfg, lnfb, hb);
  gemm(hb, Wh, bh, nullptr, logits, Vc, Dc, 0, 0);
  gpt_zero_kernel<<<1, 32, 0, stream>>>(loss);
  gpt_loss_kernel<<<BTc, 256, 0, stream>>>(logits, tgt, loss);
}
